// MultiHeadMaskedSelfAttention_52948356825791
// MI455X (gfx1250) — matrix operand order repaired and checked by reading, not compiled, not run on hardware
//
#include <hip/hip_runtime.h>
#include <hip/hip_bf16.h>

#define HDIM 1024
#define NH   16
#define HEAD 64
#define TSEQ 2048
#define BATCH 2

typedef __attribute__((ext_vector_type(16))) __bf16       v16bf;
typedef __attribute__((ext_vector_type(8)))  float        v8f;
typedef __attribute__((ext_vector_type(4)))  unsigned int u32x4;
typedef __attribute__((ext_vector_type(4)))  int          v4i;

#define AS1 __attribute__((address_space(1)))
#define AS3 __attribute__((address_space(3)))

#if defined(__gfx1250__) && __has_builtin(__builtin_amdgcn_global_load_async_to_lds_b128)
#define HAVE_ASYNC_LDS 1
#else
#define HAVE_ASYNC_LDS 0
#endif

union FragBF { v16bf v; u32x4 q[2]; };
union Pack8  { u32x4 q; __bf16 h[8]; };

// Load a 16-element bf16 WMMA fragment as two 16-byte chunks (halfs offsets o0,o1).
static __device__ inline v16bf ld_frag(const __bf16* p, int o0, int o1) {
  FragBF f;
  f.q[0] = *reinterpret_cast<const u32x4*>(p + o0);
  f.q[1] = *reinterpret_cast<const u32x4*>(p + o1);
  return f.v;
}

static __device__ inline v8f wmma_bf16(v16bf a, v16bf b, v8f c) {
  return __builtin_amdgcn_wmma_f32_16x16x32_bf16(false, a, false, b, (short)0, c,
                                                 false, false);
}

// Async 16B global -> LDS copy (ASYNCcnt-tracked); falls back to a
// register-staged copy when the builtin is unavailable.
static __device__ inline void async_copy_b128(const __bf16* g, __bf16* l) {
#if HAVE_ASYNC_LDS
  __builtin_amdgcn_global_load_async_to_lds_b128((AS1 v4i*)g, (AS3 v4i*)l, 0, 0);
#else
  *reinterpret_cast<u32x4*>(l) = *reinterpret_cast<const u32x4*>(g);
#endif
}

static __device__ inline void async_drain() {
#if HAVE_ASYNC_LDS
  asm volatile("s_wait_asynccnt 0" ::: "memory");
#endif
}

// ---------------------------------------------------------------------------
// Kernel 1: P = x @ W^T + b  (W in {Wq,Wk,Wv} by blockIdx.z), bf16 out in
// head-split layout [B, NH, T, HEAD]. Q additionally scaled by 1/sqrt(HDIM).
// Block: 256 thr (8 waves); tile 128(M) x 128(N); K staged 64 at a time with
// register double-buffering so global loads overlap the WMMA stream.
// ---------------------------------------------------------------------------
__global__ __launch_bounds__(256)
void qkv_proj_kernel(const float* __restrict__ x,
                     const float* __restrict__ Wq, const float* __restrict__ bq,
                     const float* __restrict__ Wk, const float* __restrict__ bk,
                     const float* __restrict__ Wv, const float* __restrict__ bv,
                     __bf16* __restrict__ Qo, __bf16* __restrict__ Ko,
                     __bf16* __restrict__ Vo)
{
  __shared__ __align__(16) __bf16 At[128 * 72];  // x tile [m][k]
  __shared__ __align__(16) __bf16 Bt[128 * 72];  // W tile [n][k]

  const int z = blockIdx.z;
  const float* W    = (z == 0) ? Wq : ((z == 1) ? Wk : Wv);
  const float* bias = (z == 0) ? bq : ((z == 1) ? bk : bv);
  __bf16*      dst  = (z == 0) ? Qo : ((z == 1) ? Ko : Vo);
  const float  oscale = (z == 0) ? 0.03125f : 1.0f;  // 1/sqrt(1024) folded into Q

  const int tid  = threadIdx.x;
  const int lane = tid & 31;
  const int wave = tid >> 5;
  const int wm   = wave & 3;   // 4 wave-rows * 32 rows
  const int wn   = wave >> 2;  // 2 wave-cols * 64 cols
  const int m0   = blockIdx.y * 128;
  const int n0   = blockIdx.x * 128;
  const int half = lane >> 4;
  const int lm   = lane & 15;

  v8f acc[2][4];
  #pragma unroll
  for (int mt = 0; mt < 2; ++mt)
    #pragma unroll
    for (int nt = 0; nt < 4; ++nt)
      #pragma unroll
      for (int r = 0; r < 8; ++r) acc[mt][nt][r] = 0.0f;

  float4 xr[8], wr[8];  // double-buffer registers for the staged tiles

  auto load_chunk = [&](int kb) {
    #pragma unroll
    for (int i = 0; i < 8; ++i) {
      const int idx = tid + i * 256;   // 0..2047
      const int row = idx >> 4;        // 0..127
      const int c4  = idx & 15;        // float4 column
      xr[i] = *reinterpret_cast<const float4*>(
          x + (size_t)(m0 + row) * HDIM + kb + c4 * 4);
      wr[i] = *reinterpret_cast<const float4*>(
          W + (size_t)(n0 + row) * HDIM + kb + c4 * 4);
    }
  };
  auto store_chunk = [&]() {
    #pragma unroll
    for (int i = 0; i < 8; ++i) {
      const int idx = tid + i * 256;
      const int row = idx >> 4;
      const int c4  = idx & 15;
      __bf16 hx[4] = {(__bf16)xr[i].x, (__bf16)xr[i].y, (__bf16)xr[i].z,
                      (__bf16)xr[i].w};
      *reinterpret_cast<unsigned long long*>(&At[row * 72 + c4 * 4]) =
          *reinterpret_cast<unsigned long long*>(hx);
      __bf16 hw[4] = {(__bf16)wr[i].x, (__bf16)wr[i].y, (__bf16)wr[i].z,
                      (__bf16)wr[i].w};
      *reinterpret_cast<unsigned long long*>(&Bt[row * 72 + c4 * 4]) =
          *reinterpret_cast<unsigned long long*>(hw);
    }
  };

  load_chunk(0);
  for (int kb = 0; kb < HDIM; kb += 64) {
    store_chunk();
    __syncthreads();
    if (kb + 64 < HDIM) load_chunk(kb + 64);  // prefetch; overlaps WMMAs below

    #pragma unroll
    for (int ks = 0; ks < 2; ++ks) {
      v16bf a[2], b[4];
      #pragma unroll
      for (int mt = 0; mt < 2; ++mt)
        a[mt] = ld_frag(&At[(wm * 32 + mt * 16 + lm) * 72],
                        ks * 32 + half * 8, ks * 32 + 16 + half * 8);
      #pragma unroll
      for (int nt = 0; nt < 4; ++nt)
        b[nt] = ld_frag(&Bt[(wn * 64 + nt * 16 + lm) * 72],
                        ks * 32 + half * 8, ks * 32 + 16 + half * 8);
      #pragma unroll
      for (int mt = 0; mt < 2; ++mt)
        #pragma unroll
        for (int nt = 0; nt < 4; ++nt)
          acc[mt][nt] = wmma_bf16(a[mt], b[nt], acc[mt][nt]);
    }
    __syncthreads();
  }

  // Bias, scale, store bf16 in [B, NH, T, HEAD].
  #pragma unroll
  for (int nt = 0; nt < 4; ++nt) {
    const int ng = n0 + wn * 64 + nt * 16 + lm;
    const float bsv = bias[ng];
    const int h = ng >> 6, d = ng & 63;
    #pragma unroll
    for (int mt = 0; mt < 2; ++mt) {
      #pragma unroll
      for (int r = 0; r < 8; ++r) {
        const int mg = m0 + wm * 32 + mt * 16 + half * 8 + r;
        const int bb = mg >> 11, t = mg & (TSEQ - 1);
        const float v = (acc[mt][nt][r] + bsv) * oscale;
        dst[(((size_t)(bb * NH + h)) * TSEQ + t) * HEAD + d] = (__bf16)v;
      }
    }
  }
}

// ---------------------------------------------------------------------------
// Kernel 2: causal flash attention per (batch*head). Block = 64 q rows,
// 4 waves x 16 rows. K chunk staged via async global->LDS (ASYNCcnt), V
// staged transposed [d][key]; online softmax; P routed through wave-private
// LDS to become WMMA A-fragments for P @ V.
// ---------------------------------------------------------------------------
__global__ __launch_bounds__(128)
void attn_kernel(const __bf16* __restrict__ Q, const __bf16* __restrict__ K,
                 const __bf16* __restrict__ V, float* __restrict__ out)
{
  __shared__ __align__(16) __bf16 Kt[64 * 72];      // [key][d]
  __shared__ __align__(16) __bf16 Vt[64 * 72];      // [d][key] (transposed)
  __shared__ __align__(16) __bf16 Pt[4 * 16 * 72];  // per-wave P tiles

  const int qb   = blockIdx.x;        // 64-row query block
  const int bh   = blockIdx.y;        // batch*head
  const int tid  = threadIdx.x;
  const int lane = tid & 31;
  const int w    = tid >> 5;
  const int half = lane >> 4;
  const int lm   = lane & 15;

  const size_t bhbase = (size_t)bh * TSEQ * HEAD;

  // Q fragments for this wave's 16 rows (Q already pre-scaled by 1/32).
  const __bf16* qrow = Q + bhbase + (size_t)(qb * 64 + w * 16 + lm) * HEAD;
  const v16bf aq0 = ld_frag(qrow, half * 8, 16 + half * 8);
  const v16bf aq1 = ld_frag(qrow, 32 + half * 8, 48 + half * 8);

  v8f o[4];
  float m_i[8], l_i[8];
  #pragma unroll
  for (int dt = 0; dt < 4; ++dt)
    #pragma unroll
    for (int r = 0; r < 8; ++r) o[dt][r] = 0.0f;
  #pragma unroll
  for (int r = 0; r < 8; ++r) { m_i[r] = -1e30f; l_i[r] = 0.0f; }

  for (int kb = 0; kb <= qb; ++kb) {
    // K chunk: pure async copy to LDS (bf16 already, row-major is the B-frag
    // layout for Q*K^T). V chunk: manual transposed staging.
    #pragma unroll
    for (int i = 0; i < 4; ++i) {
      const int idx = tid + i * 128;
      const int row = idx >> 3;  // key index 0..63
      const int c   = idx & 7;   // 8-half column group
      async_copy_b128(K + bhbase + (size_t)(kb * 64 + row) * HEAD + c * 8,
                      &Kt[row * 72 + c * 8]);
      Pack8 pv;
      pv.q = *reinterpret_cast<const u32x4*>(
          V + bhbase + (size_t)(kb * 64 + row) * HEAD + c * 8);
      #pragma unroll
      for (int j = 0; j < 8; ++j) Vt[(c * 8 + j) * 72 + row] = pv.h[j];
    }
    async_drain();      // each wave drains its own ASYNCcnt, then barrier
    __syncthreads();

    // S = Q K^T for this wave's 16 rows x 64 keys.
    v8f sc[4];
    #pragma unroll
    for (int nt = 0; nt < 4; ++nt) {
      const __bf16* kr = &Kt[(nt * 16 + lm) * 72];
      const v16bf b0 = ld_frag(kr, half * 8, 16 + half * 8);
      const v16bf b1 = ld_frag(kr, 32 + half * 8, 48 + half * 8);
      v8f c0;
      #pragma unroll
      for (int r = 0; r < 8; ++r) c0[r] = 0.0f;
      c0 = wmma_bf16(aq0, b0, c0);
      sc[nt] = wmma_bf16(aq1, b1, c0);
    }

    // Causal mask, diagonal chunk only.
    if (kb == qb) {
      #pragma unroll
      for (int nt = 0; nt < 4; ++nt) {
        const int ki = kb * 64 + nt * 16 + lm;
        #pragma unroll
        for (int r = 0; r < 8; ++r) {
          const int qi = qb * 64 + w * 16 + half * 8 + r;
          if (ki > qi) sc[nt][r] = -1e30f;
        }
      }
    }

    // Online softmax (rows live across the 16 lanes of each half-wave).
    float corr[8];
    #pragma unroll
    for (int r = 0; r < 8; ++r) {
      float rm = fmaxf(fmaxf(sc[0][r], sc[1][r]), fmaxf(sc[2][r], sc[3][r]));
      rm = fmaxf(rm, __shfl_xor(rm, 1));
      rm = fmaxf(rm, __shfl_xor(rm, 2));
      rm = fmaxf(rm, __shfl_xor(rm, 4));
      rm = fmaxf(rm, __shfl_xor(rm, 8));
      const float mnew = fmaxf(m_i[r], rm);
      const float cr = __expf(m_i[r] - mnew);
      m_i[r] = mnew;
      float s = 0.0f;
      #pragma unroll
      for (int nt = 0; nt < 4; ++nt) {
        const float p = __expf(sc[nt][r] - mnew);
        sc[nt][r] = p;
        s += p;
      }
      s += __shfl_xor(s, 1);
      s += __shfl_xor(s, 2);
      s += __shfl_xor(s, 4);
      s += __shfl_xor(s, 8);
      l_i[r] = l_i[r] * cr + s;
      corr[r] = cr;
    }
    #pragma unroll
    for (int dt = 0; dt < 4; ++dt)
      #pragma unroll
      for (int r = 0; r < 8; ++r) o[dt][r] *= corr[r];

    // P -> wave-private LDS (C-layout -> A-layout transpose via memory).
    __bf16* pb = &Pt[w * 16 * 72];
    #pragma unroll
    for (int nt = 0; nt < 4; ++nt)
      #pragma unroll
      for (int r = 0; r < 8; ++r)
        pb[(half * 8 + r) * 72 + nt * 16 + lm] = (__bf16)sc[nt][r];
    asm volatile("s_wait_dscnt 0" ::: "memory");

    // O += P @ V
    #pragma unroll
    for (int kh = 0; kh < 2; ++kh) {
      const v16bf ap = ld_frag(pb + lm * 72, kh * 32 + half * 8,
                               kh * 32 + 16 + half * 8);
      #pragma unroll
      for (int dt = 0; dt < 4; ++dt) {
        const __bf16* vr = &Vt[(dt * 16 + lm) * 72];
        const v16bf bvf = ld_frag(vr, kh * 32 + half * 8,
                                  kh * 32 + 16 + half * 8);
        o[dt] = wmma_bf16(ap, bvf, o[dt]);
      }
    }
    __syncthreads();
  }

  // Normalize and store fp32 output [B, T, HDIM].
  const int b = bh >> 4, h = bh & 15;
  #pragma unroll
  for (int dt = 0; dt < 4; ++dt) {
    #pragma unroll
    for (int r = 0; r < 8; ++r) {
      const int t = qb * 64 + w * 16 + half * 8 + r;
      out[((size_t)(b * TSEQ + t)) * HDIM + h * HEAD + dt * 16 + lm] =
          o[dt][r] / l_i[r];
    }
  }
}

extern "C" void kernel_launch(void* const* d_in, const int* in_sizes, int n_in,
                              void* d_out, int out_size, void* d_ws, size_t ws_size,
                              hipStream_t stream) {
  const float* x  = (const float*)d_in[0];
  const float* Wq = (const float*)d_in[1];
  const float* bq = (const float*)d_in[2];
  const float* Wk = (const float*)d_in[3];
  const float* bk = (const float*)d_in[4];
  const float* Wv = (const float*)d_in[5];
  const float* bv = (const float*)d_in[6];

  const size_t nElem = (size_t)BATCH * NH * TSEQ * HEAD;  // 4,194,304 bf16 each
  __bf16* Qw = (__bf16*)d_ws;
  __bf16* Kw = Qw + nElem;
  __bf16* Vw = Kw + nElem;

  qkv_proj_kernel<<<dim3(HDIM / 128, (BATCH * TSEQ) / 128, 3), 256, 0, stream>>>(
      x, Wq, bq, Wk, bk, Wv, bv, Qw, Kw, Vw);
  attn_kernel<<<dim3(TSEQ / 64, BATCH * NH), 128, 0, stream>>>(
      Qw, Kw, Vw, (float*)d_out);
}
